// TransformerBlock_50268297232893
// MI455X (gfx1250) — matrix operand order repaired and checked by reading, not compiled, not run on hardware
//
#include <hip/hip_runtime.h>
#include <hip/hip_bf16.h>

typedef _Float16 half_t;
typedef __attribute__((ext_vector_type(16))) _Float16 v16h;
typedef __attribute__((ext_vector_type(8)))  float    v8f;
typedef __attribute__((ext_vector_type(4)))  float    f32x4;

union FragA { v16h v; _Float16 h[16]; f32x4 q[2]; };
union FragC { v8f  v; float     f[8]; };

#define TB_L  2048
#define TB_D  1024
#define TB_H  16
#define TB_F  4096
#define TB_DH 64
#define TB_NEG (-1.0e9f)
#define TB_EPS (1e-6f)

// ---------------------------------------------------------------------------
// fp32 -> fp16 elementwise conversion (for A-side operands)
// ---------------------------------------------------------------------------
__global__ void cvt_f32_f16_kernel(const float* __restrict__ src,
                                   half_t* __restrict__ dst, int n) {
  int i = blockIdx.x * blockDim.x + threadIdx.x;
  if (i < n) dst[i] = (half_t)src[i];
}

// ---------------------------------------------------------------------------
// fp32 -> fp16 transpose-convert: src[K][N] -> dst[N][K]  (for B-side weights)
// ---------------------------------------------------------------------------
__global__ void cvt_transpose_kernel(const float* __restrict__ src,
                                     half_t* __restrict__ dst,
                                     int K, int N) {
  __shared__ float tile[32][33];
  const int k0 = blockIdx.y * 32;
  const int n0 = blockIdx.x * 32;
  const int tx = threadIdx.x;
#pragma unroll
  for (int j = threadIdx.y; j < 32; j += 8)
    tile[j][tx] = src[(size_t)(k0 + j) * N + n0 + tx];
  __syncthreads();
#pragma unroll
  for (int j = threadIdx.y; j < 32; j += 8)
    dst[(size_t)(n0 + j) * K + k0 + tx] = (half_t)tile[tx][j];
}

// ---------------------------------------------------------------------------
// Tiled WMMA GEMM:  C[M,N] = (A[M,K] x Bt[N,K]^T + bias[N]) * outScale
// Block: 256 threads = 8 waves, WG tile 128x128, wave tile 32x64
// (2x4 WMMA 16x16), K step 32, register double-buffered staging.
// OUTMODE: 0 = f32 [M,N], 1 = f16 [M,N], 2 = f16 transposed [N,M].
// ---------------------------------------------------------------------------
template <int OUTMODE, bool RELU>
__launch_bounds__(256)
__global__ void gemm_wmma_kernel(const half_t* __restrict__ A,
                                 const half_t* __restrict__ Bt,
                                 const float* __restrict__ bias,
                                 void* __restrict__ Cout,
                                 int M, int N, int K, float outScale) {
  __shared__ __align__(16) half_t As[2][128][40];  // row = 80B (16B aligned)
  __shared__ __align__(16) half_t Bs[2][128][40];

  const int tid   = threadIdx.x;
  const int wave  = tid >> 5;
  const int lane  = tid & 31;
  const int lhalf = lane & 15;
  const int hi    = lane >> 4;
  const int wm    = wave >> 1;   // 0..3 -> 32-row strip
  const int wn    = wave & 1;    // 0..1 -> 64-col strip
  const int m0    = blockIdx.y * 128;
  const int n0    = blockIdx.x * 128;

  const int sr = tid >> 1;
  const int sc = (tid & 1) * 16;
  const half_t* aSrc = A  + (size_t)(m0 + sr) * K + sc;
  const half_t* bSrc = Bt + (size_t)(n0 + sr) * K + sc;

  f32x4 ra0, ra1, rb0, rb1;
  FragC acc[2][4];
#pragma unroll
  for (int a = 0; a < 2; ++a)
#pragma unroll
    for (int b = 0; b < 4; ++b) acc[a][b].v = (v8f){};

  const int nt = K >> 5;

  {
    ra0 = *(const f32x4*)(aSrc);  ra1 = *(const f32x4*)(aSrc + 8);
    rb0 = *(const f32x4*)(bSrc);  rb1 = *(const f32x4*)(bSrc + 8);
    *(f32x4*)&As[0][sr][sc] = ra0;  *(f32x4*)&As[0][sr][sc + 8] = ra1;
    *(f32x4*)&Bs[0][sr][sc] = rb0;  *(f32x4*)&Bs[0][sr][sc + 8] = rb1;
  }
  __syncthreads();

  for (int kt = 0; kt < nt; ++kt) {
    const int cur = kt & 1;
    if (kt + 1 < nt) {
      const half_t* a = aSrc + (size_t)(kt + 1) * 32;
      const half_t* b = bSrc + (size_t)(kt + 1) * 32;
      ra0 = *(const f32x4*)(a);  ra1 = *(const f32x4*)(a + 8);
      rb0 = *(const f32x4*)(b);  rb1 = *(const f32x4*)(b + 8);
    }
    if (kt + 2 < nt) {  // L2 prefetch two tiles ahead (global_prefetch_b8)
      __builtin_prefetch(aSrc + (size_t)(kt + 2) * 32, 0, 0);
      __builtin_prefetch(bSrc + (size_t)(kt + 2) * 32, 0, 0);
    }

    FragA af[2];
#pragma unroll
    for (int fi = 0; fi < 2; ++fi) {
      const int r = wm * 32 + fi * 16 + lhalf;
      af[fi].q[0] = *(const f32x4*)&As[cur][r][hi * 16];
      af[fi].q[1] = *(const f32x4*)&As[cur][r][hi * 16 + 8];
    }
#pragma unroll
    for (int fj = 0; fj < 4; ++fj) {
      FragA bf;
      const int c = wn * 64 + fj * 16 + lhalf;
      bf.q[0] = *(const f32x4*)&Bs[cur][c][hi * 16];
      bf.q[1] = *(const f32x4*)&Bs[cur][c][hi * 16 + 8];
#pragma unroll
      for (int fi = 0; fi < 2; ++fi)
        acc[fi][fj].v = __builtin_amdgcn_wmma_f32_16x16x32_f16(
            false, af[fi].v, false, bf.v, (short)0, acc[fi][fj].v, false, false);
    }

    if (kt + 1 < nt) {
      const int nxt = cur ^ 1;
      *(f32x4*)&As[nxt][sr][sc] = ra0;  *(f32x4*)&As[nxt][sr][sc + 8] = ra1;
      *(f32x4*)&Bs[nxt][sr][sc] = rb0;  *(f32x4*)&Bs[nxt][sr][sc + 8] = rb1;
    }
    __syncthreads();
  }

#pragma unroll
  for (int fi = 0; fi < 2; ++fi)
#pragma unroll
    for (int fj = 0; fj < 4; ++fj) {
      const int n  = n0 + wn * 64 + fj * 16 + lhalf;
      const float bv = bias ? bias[n] : 0.0f;
#pragma unroll
      for (int i = 0; i < 8; ++i) {
        const int m = m0 + wm * 32 + fi * 16 + hi * 8 + i;
        float v = (acc[fi][fj].f[i] + bv) * outScale;
        if (RELU) v = fmaxf(v, 0.0f);
        if (OUTMODE == 0)      ((float*)Cout)[(size_t)m * N + n]  = v;
        else if (OUTMODE == 1) ((half_t*)Cout)[(size_t)m * N + n] = (half_t)v;
        else                   ((half_t*)Cout)[(size_t)n * M + m] = (half_t)v;
      }
    }
}

// ---------------------------------------------------------------------------
// Flash attention with WMMA.
// Q is pre-scaled by 1/sqrt(DH) at projection time, so S = Qs * K^T directly.
// Q/K stored [L, D] f16; V stored transposed [D, L] f16.
// Grid: (L/128, H). Block: 128 threads = 4 waves; each wave owns 2 query
// tiles (32 rows). Mask (i >= j -> +(-1e9)) is applied with wave-uniform
// block classification: most key blocks are fully masked (uniform add) or
// fully unmasked (no-op); at most one boundary block per tile needs the
// per-element compare path.
// ---------------------------------------------------------------------------
__launch_bounds__(128)
__global__ void attn_wmma_kernel(const half_t* __restrict__ Qh,
                                 const half_t* __restrict__ Kh,
                                 const half_t* __restrict__ Vt,
                                 half_t* __restrict__ ctx) {
  const int h     = blockIdx.y;
  const int wid   = threadIdx.x >> 5;
  const int lane  = threadIdx.x & 31;
  const int lhalf = lane & 15;
  const int hi    = lane >> 4;
  const int q0    = blockIdx.x * 128 + wid * 32;
  const int hoff  = h * TB_DH;

  __shared__ __align__(16) half_t Psh[4][2][16][32];

  FragA qf[2][2];
#pragma unroll
  for (int qt = 0; qt < 2; ++qt)
#pragma unroll
    for (int kf = 0; kf < 2; ++kf) {
      const int r    = q0 + qt * 16 + lhalf;
      const int base = hoff + kf * 32;
      qf[qt][kf].q[0] = *(const f32x4*)(Qh + (size_t)r * TB_D + base + hi * 16);
      qf[qt][kf].q[1] = *(const f32x4*)(Qh + (size_t)r * TB_D + base + hi * 16 + 8);
    }

  FragC o[2][4];
#pragma unroll
  for (int qt = 0; qt < 2; ++qt)
#pragma unroll
    for (int t = 0; t < 4; ++t) o[qt][t].v = (v8f){};
  float mrow[2][8], lrow[2][8];
#pragma unroll
  for (int qt = 0; qt < 2; ++qt)
#pragma unroll
    for (int i = 0; i < 8; ++i) { mrow[qt][i] = -3.0e38f; lrow[qt][i] = 0.0f; }

  for (int j0 = 0; j0 < TB_L; j0 += 32) {
    // ---- K^T fragments (shared by both query tiles) ----
    FragA kfr[2][2];
#pragma unroll
    for (int jj = 0; jj < 2; ++jj)
#pragma unroll
      for (int kf = 0; kf < 2; ++kf) {
        const int key = j0 + jj * 16 + lhalf;
        const half_t* p = Kh + (size_t)key * TB_D + hoff + kf * 32 + hi * 16;
        kfr[jj][kf].q[0] = *(const f32x4*)(p);
        kfr[jj][kf].q[1] = *(const f32x4*)(p + 8);
      }

#pragma unroll
    for (int qt = 0; qt < 2; ++qt) {
      FragC s[2];
#pragma unroll
      for (int jj = 0; jj < 2; ++jj) {
        s[jj].v = (v8f){};
#pragma unroll
        for (int kf = 0; kf < 2; ++kf)
          s[jj].v = __builtin_amdgcn_wmma_f32_16x16x32_f16(
              false, qf[qt][kf].v, false, kfr[jj][kf].v, (short)0, s[jj].v,
              false, false);
      }

      // ---- mask: wave-uniform block classification ----
      const int mbase = q0 + qt * 16;     // rows mbase .. mbase+15
      if (j0 + 31 <= mbase) {
        // fully masked block: uniform shift
#pragma unroll
        for (int jj = 0; jj < 2; ++jj)
#pragma unroll
          for (int i = 0; i < 8; ++i) s[jj].f[i] += TB_NEG;
      } else if (j0 < mbase + 16) {
        // boundary block: per-element mask
#pragma unroll
        for (int i = 0; i < 8; ++i) {
          const int m = mbase + hi * 8 + i;
#pragma unroll
          for (int jj = 0; jj < 2; ++jj) {
            const int n = j0 + jj * 16 + lhalf;
            if (m >= n) s[jj].f[i] += TB_NEG;
          }
        }
      }
      // else: fully unmasked, nothing to do

      // ---- online softmax (row = 8*hi + slot) ----
      float mnew[8];
#pragma unroll
      for (int i = 0; i < 8; ++i) {
        float best = fmaxf(s[0].f[i], s[1].f[i]);
#pragma unroll
        for (int off = 1; off < 16; off <<= 1)
          best = fmaxf(best, __shfl_xor(best, off, 32));
        mnew[i] = fmaxf(mrow[qt][i], best);
      }
#pragma unroll
      for (int i = 0; i < 8; ++i) {
        const float corr = __expf(mrow[qt][i] - mnew[i]);
        mrow[qt][i] = mnew[i];
        lrow[qt][i] *= corr;
#pragma unroll
        for (int t = 0; t < 4; ++t) o[qt][t].f[i] *= corr;
      }
#pragma unroll
      for (int jj = 0; jj < 2; ++jj)
#pragma unroll
        for (int i = 0; i < 8; ++i) {
          float p = __expf(s[jj].f[i] - mrow[qt][i]);
          float ps = p;
#pragma unroll
          for (int off = 1; off < 16; off <<= 1) ps += __shfl_xor(ps, off, 32);
          lrow[qt][i] += ps;
          Psh[wid][qt][hi * 8 + i][jj * 16 + lhalf] = (half_t)p;
        }
    }
    __syncthreads();

    // ---- O += P * V ----
    FragA pf[2];
#pragma unroll
    for (int qt = 0; qt < 2; ++qt) {
      pf[qt].q[0] = *(const f32x4*)&Psh[wid][qt][lhalf][hi * 16];
      pf[qt].q[1] = *(const f32x4*)&Psh[wid][qt][lhalf][hi * 16 + 8];
    }
#pragma unroll
    for (int t = 0; t < 4; ++t) {
      FragA vf;
      const int row = hoff + t * 16 + lhalf;
      const half_t* p = Vt + (size_t)row * TB_L + j0 + hi * 16;
      vf.q[0] = *(const f32x4*)(p);
      vf.q[1] = *(const f32x4*)(p + 8);
#pragma unroll
      for (int qt = 0; qt < 2; ++qt)
        o[qt][t].v = __builtin_amdgcn_wmma_f32_16x16x32_f16(
            false, pf[qt].v, false, vf.v, (short)0, o[qt][t].v, false, false);
    }
    __syncthreads();
  }

  // ---- normalize (one reciprocal per row slot), store ctx [L, D] f16 ----
#pragma unroll
  for (int qt = 0; qt < 2; ++qt) {
#pragma unroll
    for (int i = 0; i < 8; ++i) lrow[qt][i] = 1.0f / lrow[qt][i];
#pragma unroll
    for (int t = 0; t < 4; ++t)
#pragma unroll
      for (int i = 0; i < 8; ++i) {
        const float val = o[qt][t].f[i] * lrow[qt][i];
        ctx[(size_t)(q0 + qt * 16 + hi * 8 + i) * TB_D + hoff + t * 16 + lhalf] =
            (half_t)val;
      }
  }
}

// ---------------------------------------------------------------------------
// out = LayerNorm(a + b) * g + be   (one 256-thread block per row, D = 1024)
// ---------------------------------------------------------------------------
__launch_bounds__(256)
__global__ void add_ln_kernel(const float* __restrict__ a,
                              const float* __restrict__ b,
                              const float* __restrict__ g,
                              const float* __restrict__ be,
                              float* __restrict__ out32,
                              half_t* __restrict__ out16) {
  const int row = blockIdx.x;
  const int t   = threadIdx.x;
  __shared__ float red[256];

  float v[4];
  float s = 0.0f;
#pragma unroll
  for (int i = 0; i < 4; ++i) {
    const int c = t + i * 256;
    const float x = a[(size_t)row * TB_D + c] + b[(size_t)row * TB_D + c];
    v[i] = x;
    s += x;
  }
  red[t] = s;
  __syncthreads();
  for (int st = 128; st > 0; st >>= 1) {
    if (t < st) red[t] += red[t + st];
    __syncthreads();
  }
  const float mu = red[0] * (1.0f / TB_D);
  __syncthreads();

  float vs = 0.0f;
#pragma unroll
  for (int i = 0; i < 4; ++i) {
    const float d = v[i] - mu;
    vs += d * d;
  }
  red[t] = vs;
  __syncthreads();
  for (int st = 128; st > 0; st >>= 1) {
    if (t < st) red[t] += red[t + st];
    __syncthreads();
  }
  const float var = red[0] * (1.0f / TB_D);
  const float r   = rsqrtf(var + TB_EPS);

#pragma unroll
  for (int i = 0; i < 4; ++i) {
    const int c = t + i * 256;
    const float y = (v[i] - mu) * r * g[c] + be[c];
    out32[(size_t)row * TB_D + c] = y;
    if (out16) out16[(size_t)row * TB_D + c] = (half_t)y;
  }
}

// ---------------------------------------------------------------------------
// Host-side orchestration
// ---------------------------------------------------------------------------
static inline void launch_cvt(const float* src, half_t* dst, int n, hipStream_t s) {
  cvt_f32_f16_kernel<<<(n + 255) / 256, 256, 0, s>>>(src, dst, n);
}
static inline void launch_cvt_t(const float* src, half_t* dst, int K, int N,
                                hipStream_t s) {
  cvt_transpose_kernel<<<dim3(N / 32, K / 32), dim3(32, 8), 0, s>>>(src, dst, K, N);
}

extern "C" void kernel_launch(void* const* d_in, const int* in_sizes, int n_in,
                              void* d_out, int out_size, void* d_ws, size_t ws_size,
                              hipStream_t stream) {
  (void)in_sizes; (void)n_in; (void)out_size; (void)ws_size;

  const float* x   = (const float*)d_in[0];
  const float* Wq  = (const float*)d_in[1];
  const float* bq  = (const float*)d_in[2];
  const float* Wk  = (const float*)d_in[3];
  const float* bk  = (const float*)d_in[4];
  const float* Wv  = (const float*)d_in[5];
  const float* bv  = (const float*)d_in[6];
  const float* Wo  = (const float*)d_in[7];
  const float* bo  = (const float*)d_in[8];
  const float* W1  = (const float*)d_in[9];
  const float* b1  = (const float*)d_in[10];
  const float* W2  = (const float*)d_in[11];
  const float* b2  = (const float*)d_in[12];
  const float* g1  = (const float*)d_in[13];
  const float* be1 = (const float*)d_in[14];
  const float* g2  = (const float*)d_in[15];
  const float* be2 = (const float*)d_in[16];

  char* ws = (char*)d_ws;
  size_t off = 0;
  auto take = [&](size_t bytes) -> void* {
    void* p = ws + off;
    off = (off + bytes + 255) & ~(size_t)255;
    return p;
  };

  const size_t LD = (size_t)TB_L * TB_D;  // 2M
  const size_t DD = (size_t)TB_D * TB_D;  // 1M
  const size_t DF = (size_t)TB_D * TB_F;  // 4M
  const size_t LF = (size_t)TB_L * TB_F;  // 8M

  half_t* xh   = (half_t*)take(LD * 2);
  half_t* Wqt  = (half_t*)take(DD * 2);
  half_t* Wkt  = (half_t*)take(DD * 2);
  half_t* Wvt  = (half_t*)take(DD * 2);
  half_t* Wot  = (half_t*)take(DD * 2);
  half_t* W1t  = (half_t*)take(DF * 2);
  half_t* W2t  = (half_t*)take(DF * 2);
  half_t* Qh   = (half_t*)take(LD * 2);
  half_t* Kh   = (half_t*)take(LD * 2);
  half_t* Vt   = (half_t*)take(LD * 2);   // V transposed: [D][L]
  half_t* ctxh = (half_t*)take(LD * 2);
  float*  attn = (float*)take(LD * 4);
  float*  ln1f = (float*)take(LD * 4);
  half_t* ln1h = (half_t*)take(LD * 2);
  half_t* hh   = (half_t*)take(LF * 2);
  float*  ffn  = (float*)take(LD * 4);

  launch_cvt(x, xh, (int)LD, stream);
  launch_cvt_t(Wq, Wqt, TB_D, TB_D, stream);
  launch_cvt_t(Wk, Wkt, TB_D, TB_D, stream);
  launch_cvt_t(Wv, Wvt, TB_D, TB_D, stream);
  launch_cvt_t(Wo, Wot, TB_D, TB_D, stream);
  launch_cvt_t(W1, W1t, TB_D, TB_F, stream);
  launch_cvt_t(W2, W2t, TB_F, TB_D, stream);

  dim3 blk(256);
  // QKV projections: Q pre-scaled by 1/sqrt(DH); V written transposed [D,L]
  {
    dim3 grd(TB_D / 128, TB_L / 128);
    gemm_wmma_kernel<1, false><<<grd, blk, 0, stream>>>(xh, Wqt, bq, (void*)Qh, TB_L, TB_D, TB_D, 0.125f);
    gemm_wmma_kernel<1, false><<<grd, blk, 0, stream>>>(xh, Wkt, bk, (void*)Kh, TB_L, TB_D, TB_D, 1.0f);
    gemm_wmma_kernel<2, false><<<grd, blk, 0, stream>>>(xh, Wvt, bv, (void*)Vt, TB_L, TB_D, TB_D, 1.0f);
  }
  {
    dim3 grd(TB_L / 128, TB_H);
    attn_wmma_kernel<<<grd, dim3(128), 0, stream>>>(Qh, Kh, Vt, ctxh);
  }
  {
    dim3 grd(TB_D / 128, TB_L / 128);
    gemm_wmma_kernel<0, false><<<grd, blk, 0, stream>>>(ctxh, Wot, bo, (void*)attn, TB_L, TB_D, TB_D, 1.0f);
  }
  add_ln_kernel<<<TB_L, 256, 0, stream>>>(x, attn, g1, be1, ln1f, ln1h);
  {
    dim3 grd(TB_F / 128, TB_L / 128);
    gemm_wmma_kernel<1, true><<<grd, blk, 0, stream>>>(ln1h, W1t, b1, (void*)hh, TB_L, TB_F, TB_D, 1.0f);
  }
  {
    dim3 grd(TB_D / 128, TB_L / 128);
    gemm_wmma_kernel<0, false><<<grd, blk, 0, stream>>>(hh, W2t, b2, (void*)ffn, TB_L, TB_D, TB_F, 1.0f);
  }
  add_ln_kernel<<<TB_L, 256, 0, stream>>>(ln1f, ffn, g2, be2, (float*)d_out, (half_t*)nullptr);
}
